// Encoder_61332132987531
// MI455X (gfx1250) — matrix operand order repaired and checked by reading, not compiled, not run on hardware
//
#include <hip/hip_runtime.h>

typedef __bf16 bf16;
typedef __attribute__((ext_vector_type(16))) __bf16 v16bf;
typedef __attribute__((ext_vector_type(8)))  __bf16 v8bf;
typedef __attribute__((ext_vector_type(8)))  float  v8f;

constexpr int T_LEN  = 512;
constexpr int BATCH  = 64;
constexpr int HID    = 512;
constexpr int KIN    = 1024;            // input width (== 2*HID for every layer)
constexpr int KTOT   = KIN + HID;       // 1536: [x | h] packed K
constexpr int GATES4 = 4 * HID;         // 2048
constexpr int NCHUNK = KTOT / 32;       // 48 K-chunks of 32

__device__ __forceinline__ float fast_sigmoid(float x) {
    return 1.0f / (1.0f + __expf(-x));
}
__device__ __forceinline__ float fast_tanh(float x) {
    return 2.0f / (1.0f + __expf(-2.0f * x)) - 1.0f;
}

// ---- CDNA5 async global->LDS copy helpers -------------------------------
__device__ __forceinline__ unsigned lds_addr_of(const void* p) {
    // generic -> LDS(addrspace 3) offset
    return (unsigned)(uintptr_t)(const __attribute__((address_space(3))) char*)p;
}
__device__ __forceinline__ void async_copy16(const void* gptr, unsigned ldsoff) {
    asm volatile("global_load_async_to_lds_b128 %0, %1, off"
                 :: "v"(ldsoff), "v"(gptr)
                 : "memory");
}
// allow the newest 2 async copies (the just-issued chunk) to stay in flight
__device__ __forceinline__ void wait_async_le2() {
    asm volatile("s_wait_asynccnt 0x2" ::: "memory");
}

// ---------------------------------------------------------------------------
// Prep kernels
// ---------------------------------------------------------------------------
__global__ void pack_weights(const float* __restrict__ Wih,
                             const float* __restrict__ Whh,
                             bf16* __restrict__ Wpk) {
    long i = (long)blockIdx.x * blockDim.x + threadIdx.x;
    const long N = 4L * GATES4 * KTOT;
    if (i >= N) return;
    int  k    = (int)(i % KTOT);
    long rowg = i / KTOT;                 // ld*2048 + row
    long ld   = rowg / GATES4;
    long row  = rowg % GATES4;
    float v;
    if (k < KIN) v = Wih[(ld * GATES4 + row) * KIN + k];
    else         v = Whh[(ld * GATES4 + row) * HID + (k - KIN)];
    Wpk[i] = (bf16)v;
}

__global__ void convert_x(const float* __restrict__ x, bf16* __restrict__ xb, long n8) {
    long i = (long)blockIdx.x * blockDim.x + threadIdx.x;
    if (i >= n8) return;
    const float4* p = (const float4*)(x + i * 8);
    float4 f0 = p[0], f1 = p[1];
    v8bf a;
    a[0] = (bf16)f0.x; a[1] = (bf16)f0.y; a[2] = (bf16)f0.z; a[3] = (bf16)f0.w;
    a[4] = (bf16)f1.x; a[5] = (bf16)f1.y; a[6] = (bf16)f1.z; a[7] = (bf16)f1.w;
    *(v8bf*)(xb + i * 8) = a;
}

__global__ void bias_sum(const float* __restrict__ bih,
                         const float* __restrict__ bhh,
                         float* __restrict__ bsum) {
    int i = blockIdx.x * blockDim.x + threadIdx.x;
    if (i < 4 * GATES4) bsum[i] = bih[i] + bhh[i];
}

__global__ void init_state(bf16* __restrict__ hb, float* __restrict__ cb) {
    int i = blockIdx.x * blockDim.x + threadIdx.x;
    if (i < 2 * 2 * BATCH * HID) hb[i] = (bf16)0.0f;
    if (i < 2 * BATCH * HID)     cb[i] = 0.0f;
}

// ---------------------------------------------------------------------------
// One LSTM timestep, both directions. Grid = 64 WGs (dir = blk>>5, 16-wide
// hidden slice = blk&31), 256 threads = 8 waves. Each WG computes a 64x64
// tile (batch x {i,f,g,o}x16 cols), K = 1536. Triple-buffered LDS fed by
// global_load_async_to_lds_b128 with 2 chunks in flight (s_wait_asynccnt 2),
// math via v_wmma_f32_16x16x32_bf16.
// ---------------------------------------------------------------------------
__global__ void __launch_bounds__(256)
lstm_step(const bf16*  __restrict__ xbf,   // layer-0 input  [B,T,KIN] bf16
          const bf16*  __restrict__ y0in,  // layer-1 input  [T,B,KIN] bf16
          const bf16*  __restrict__ Wpk,   // [4][GATES4][KTOT] bf16
          const float* __restrict__ bsum,  // [4][GATES4]
          bf16*        hb,                 // [2dir][2pp][B*H] bf16
          float*       cb,                 // [2dir][B*H] f32
          bf16*        y0out,              // layer-0 output [T,B,2H] bf16
          float*       out,                // final output   [B,T,2H] f32
          int layer, int s) {
    __shared__ __align__(16) bf16  Atile[3][BATCH][40];   // 64 x 32 (+pad)
    __shared__ __align__(16) bf16  Btile[3][64][40];      // 64 cols x 32 K (+pad)
    __shared__               float gateLds[4][BATCH][16]; // i,f,g,o pre-activations

    const int tid   = threadIdx.x;
    const int dir   = blockIdx.x >> 5;
    const int slice = blockIdx.x & 31;
    const int j0    = slice * 16;
    const int t     = dir ? (T_LEN - 1 - s) : s;
    const int pp    = s & 1;

    const bf16*  Wd     = Wpk  + (size_t)(layer * 2 + dir) * GATES4 * KTOT;
    const float* bs     = bsum + (layer * 2 + dir) * GATES4;
    const bf16*  hread  = hb + (size_t)(dir * 2 + pp)       * (BATCH * HID);
    bf16*        hwrite = hb + (size_t)(dir * 2 + (pp ^ 1)) * (BATCH * HID);
    float*       cdir   = cb + (size_t)dir * (BATCH * HID);

    // cooperative-load indices: 256 thr -> 64 rows x 4 quarters of 8 bf16 (16B)
    const int lr = tid >> 2;              // 0..63 (A: batch row / B: tile col)
    const int lk = (tid & 3) * 8;         // 0,8,16,24
    const int wgate = lr >> 4;            // B col -> gate
    const int wj    = lr & 15;

    // per-thread source pointers (chunk k0 added per iteration)
    const bf16* wptr = Wd + (size_t)(wgate * HID + j0 + wj) * KTOT + lk;
    const bf16* xptr;                     // x-region rows
    if (layer == 0) xptr = xbf  + ((size_t)lr * T_LEN + t) * KIN + lk;  // [B,T,K]
    else            xptr = y0in + ((size_t)t * BATCH + lr) * KIN + lk;  // [T,B,K]
    const bf16* hptr = hread + (size_t)lr * HID + lk;                   // h-region

    // wave/fragment indices (wave32)
    const int wave = tid >> 5;
    const int lane = tid & 31;
    const int half = lane >> 4;
    const int lm   = lane & 15;
    const int m16  = (wave & 3) * 16;     // M tile (batch rows)
    const int gp   = wave >> 2;           // gate pair {2gp, 2gp+1}

    v8f acc0 = {0.f, 0.f, 0.f, 0.f, 0.f, 0.f, 0.f, 0.f};
    v8f acc1 = acc0;

    auto issue_chunk = [&](int kt, int buf) {
        const int k0 = kt * 32;
        async_copy16(wptr + k0, lds_addr_of(&Btile[buf][lr][lk]));
        const bf16* ap = (k0 < KIN) ? (xptr + k0) : (hptr + (k0 - KIN));
        async_copy16(ap, lds_addr_of(&Atile[buf][lr][lk]));
    };

    // prologue: 2 chunks in flight
    issue_chunk(0, 0);
    issue_chunk(1, 1);
    wait_async_le2();      // chunk 0 landed (chunk 1 may still be in flight)
    __syncthreads();

#pragma unroll 3
    for (int kt = 0; kt < NCHUNK; ++kt) {
        const int cur = kt % 3;
        if (kt + 2 < NCHUNK) issue_chunk(kt + 2, (kt + 2) % 3);

        // ---- fragments per ISA 16-bit layouts ----
        // A 16x32: lanes 0-15 K {0..7,16..23}, lanes 16-31 K {8..15,24..31}
        union { v16bf v; v8bf h[2]; } afrag;
        afrag.h[0] = *(const v8bf*)&Atile[cur][m16 + lm][half * 8];
        afrag.h[1] = *(const v8bf*)&Atile[cur][m16 + lm][16 + half * 8];
        // B 32x16: lane = col, lanes 0-15 K {0..7,16..23}, lanes 16-31 K {8..15,24..31}
        union { v16bf v; v8bf h[2]; } bf0, bf1;
        bf0.h[0] = *(const v8bf*)&Btile[cur][(2 * gp) * 16 + lm][half * 8];
        bf0.h[1] = *(const v8bf*)&Btile[cur][(2 * gp) * 16 + lm][16 + half * 8];
        bf1.h[0] = *(const v8bf*)&Btile[cur][(2 * gp + 1) * 16 + lm][half * 8];
        bf1.h[1] = *(const v8bf*)&Btile[cur][(2 * gp + 1) * 16 + lm][16 + half * 8];

        acc0 = __builtin_amdgcn_wmma_f32_16x16x32_bf16(false, afrag.v, false, bf0.v,
                                                       (short)0, acc0, false, false);
        acc1 = __builtin_amdgcn_wmma_f32_16x16x32_bf16(false, afrag.v, false, bf1.v,
                                                       (short)0, acc1, false, false);

        wait_async_le2();  // chunk kt+1 landed; kt+2's 2 copies may remain
        __syncthreads();   // visible to all waves; all reads of 'cur' done
    }

    // ---- bias + spill gate pre-activations to LDS ----
    {
        float bv0 = bs[(2 * gp) * HID + j0 + lm];
        float bv1 = bs[(2 * gp + 1) * HID + j0 + lm];
#pragma unroll
        for (int r = 0; r < 8; ++r) {
            int m = m16 + half * 8 + r;   // D layout: lanes16-31 hold M=8..15
            gateLds[2 * gp][m][lm]     = acc0[r] + bv0;
            gateLds[2 * gp + 1][m][lm] = acc1[r] + bv1;
        }
    }
    __syncthreads();

    // ---- elementwise LSTM cell: 64 batch x 16 hidden = 1024 items ----
#pragma unroll
    for (int q = 0; q < 4; ++q) {
        int item = tid + q * 256;
        int b = item >> 4;
        int j = item & 15;
        float iv = fast_sigmoid(gateLds[0][b][j]);
        float fv = fast_sigmoid(gateLds[1][b][j]);
        float gv = fast_tanh   (gateLds[2][b][j]);
        float ov = fast_sigmoid(gateLds[3][b][j]);
        size_t ci = (size_t)b * HID + j0 + j;
        float cn = fv * cdir[ci] + iv * gv;
        cdir[ci] = cn;
        float hn = ov * fast_tanh(cn);
        hwrite[ci] = (bf16)hn;
        if (layer == 0) {
            y0out[((size_t)t * BATCH + b) * KIN + dir * HID + j0 + j] = (bf16)hn;
        } else {
            out[((size_t)b * T_LEN + t) * KIN + dir * HID + j0 + j] = hn;
        }
    }
}

// ---------------------------------------------------------------------------
extern "C" void kernel_launch(void* const* d_in, const int* in_sizes, int n_in,
                              void* d_out, int out_size, void* d_ws, size_t ws_size,
                              hipStream_t stream) {
    const float* x   = (const float*)d_in[0];  // [64,512,1024]
    const float* Wih = (const float*)d_in[1];  // [2,2,2048,1024]
    const float* Whh = (const float*)d_in[2];  // [2,2,2048,512]
    const float* bih = (const float*)d_in[3];  // [2,2,2048]
    const float* bhh = (const float*)d_in[4];  // [2,2,2048]
    float* out = (float*)d_out;                // [64,512,1024]

    char* ws = (char*)d_ws;
    size_t off = 0;
    auto take = [&](size_t bytes) -> char* {
        char* p = ws + off;
        off += (bytes + 255) & ~(size_t)255;
        return p;
    };
    bf16*  Wpk  = (bf16*) take(4UL * GATES4 * KTOT * sizeof(bf16));         // ~25 MB
    float* bsum = (float*)take(4UL * GATES4 * sizeof(float));
    bf16*  xbf  = (bf16*) take((size_t)BATCH * T_LEN * KIN * sizeof(bf16)); // 67 MB
    bf16*  y0   = (bf16*) take((size_t)T_LEN * BATCH * KIN * sizeof(bf16)); // 67 MB
    bf16*  hb   = (bf16*) take(4UL * BATCH * HID * sizeof(bf16));
    float* cbuf = (float*)take(2UL * BATCH * HID * sizeof(float));

    {
        long n = 4L * GATES4 * KTOT;
        pack_weights<<<(int)((n + 255) / 256), 256, 0, stream>>>(Wih, Whh, Wpk);
    }
    {
        long n8 = (long)BATCH * T_LEN * KIN / 8;
        convert_x<<<(int)((n8 + 255) / 256), 256, 0, stream>>>(x, xbf, n8);
    }
    bias_sum<<<(4 * GATES4 + 255) / 256, 256, 0, stream>>>(bih, bhh, bsum);

    for (int layer = 0; layer < 2; ++layer) {
        init_state<<<(2 * 2 * BATCH * HID + 255) / 256, 256, 0, stream>>>(hb, cbuf);
        for (int s = 0; s < T_LEN; ++s) {
            lstm_step<<<64, 256, 0, stream>>>(xbf, y0, Wpk, bsum, hb, cbuf,
                                              y0, out, layer, s);
        }
    }
}
